// LlamaAttentionSparse_45165876084769
// MI455X (gfx1250) — matrix operand order repaired and checked by reading, not compiled, not run on hardware
//
#include <hip/hip_runtime.h>
#include <stdint.h>

// ---------------- CDNA5 WMMA types ----------------
typedef __attribute__((ext_vector_type(16))) __bf16 v16bf;
typedef __attribute__((ext_vector_type(8)))  float  v8f;

#define S_LEN   2048
#define DMODEL  4096
#define NHEADS  32
#define HDIM    128
#define HEAVY   256
#define RECENT  256
#define SELW    (S_LEN - RECENT)   // 1792 selectable keys
#define MASKW   (S_LEN + 1)        // 2049

struct Frag32B { uint4 a, b; };    // 32 bytes == one v16bf

static __device__ inline uint16_t f2bf(float x) {
  uint32_t u = __builtin_bit_cast(uint32_t, x);
  u = (u + 0x7FFFu + ((u >> 16) & 1u)) >> 16;   // round-nearest-even
  return (uint16_t)u;
}
static __device__ inline uint32_t pack2bf(float x, float y) {
  return (uint32_t)f2bf(x) | ((uint32_t)f2bf(y) << 16);
}
static __device__ inline float bf2f(uint16_t h) {
  return __builtin_bit_cast(float, (uint32_t)h << 16);
}

static __device__ inline v8f wmma_bf16(v16bf a, v16bf b, v8f c) {
  // D = A(16x32 bf16) * B(32x16 bf16) + C(16x16 f32)
  return __builtin_amdgcn_wmma_f32_16x16x32_bf16(
      /*neg_a=*/false, a, /*neg_b=*/false, b,
      /*c_mod=*/(short)0, c, /*reuse_a=*/false, /*reuse_b=*/false);
}

// ---- Fragment loaders (contiguous k order, same for A and B) ----
// A (16x32, 16-bit): lane L -> row M = L%16; element i: K = kk + (L/16)*16 + i
static __device__ inline v16bf frag_a_bf16(const uint16_t* __restrict__ A,
                                           int lda, int m0, int kk) {
  int lane = threadIdx.x & 31;
  const uint16_t* row = A + (size_t)(m0 + (lane & 15)) * lda;
  int kb = kk + ((lane >> 4) << 4);
  Frag32B r;
  r.a = *(const uint4*)(row + kb);
  r.b = *(const uint4*)(row + kb + 8);
  return __builtin_bit_cast(v16bf, r);
}
static __device__ inline v16bf frag_a_f32(const float* __restrict__ A,
                                          int lda, int m0, int kk) {
  int lane = threadIdx.x & 31;
  const float* row = A + (size_t)(m0 + (lane & 15)) * lda;
  int kb = kk + ((lane >> 4) << 4);
  float4 a0 = *(const float4*)(row + kb);
  float4 a1 = *(const float4*)(row + kb + 4);
  float4 b0 = *(const float4*)(row + kb + 8);
  float4 b1 = *(const float4*)(row + kb + 12);
  Frag32B r;
  r.a = make_uint4(pack2bf(a0.x, a0.y), pack2bf(a0.z, a0.w),
                   pack2bf(a1.x, a1.y), pack2bf(a1.z, a1.w));
  r.b = make_uint4(pack2bf(b0.x, b0.y), pack2bf(b0.z, b0.w),
                   pack2bf(b1.x, b1.y), pack2bf(b1.z, b1.w));
  return __builtin_bit_cast(v16bf, r);
}
// B (32x16, 16-bit): lane L -> col N = L%16; element i: K = kk + (L/16)*16 + i
// Source is stored "N-major rows x K columns" (i.e. W[n][k] or K[key][d]).
static __device__ inline v16bf frag_b_bf16(const uint16_t* __restrict__ B,
                                           int ldb, int n0, int kk) {
  int lane = threadIdx.x & 31;
  const uint16_t* row = B + (size_t)(n0 + (lane & 15)) * ldb;
  int kb = kk + ((lane >> 4) << 4);
  Frag32B r;
  r.a = *(const uint4*)(row + kb);
  r.b = *(const uint4*)(row + kb + 8);
  return __builtin_bit_cast(v16bf, r);
}
static __device__ inline v16bf frag_b_f32(const float* __restrict__ B,
                                          int ldb, int n0, int kk) {
  int lane = threadIdx.x & 31;
  const float* row = B + (size_t)(n0 + (lane & 15)) * ldb;
  int kb = kk + ((lane >> 4) << 4);
  float4 a0 = *(const float4*)(row + kb);
  float4 a1 = *(const float4*)(row + kb + 4);
  float4 a2 = *(const float4*)(row + kb + 8);
  float4 a3 = *(const float4*)(row + kb + 12);
  Frag32B r;
  r.a = make_uint4(pack2bf(a0.x, a0.y), pack2bf(a0.z, a0.w),
                   pack2bf(a1.x, a1.y), pack2bf(a1.z, a1.w));
  r.b = make_uint4(pack2bf(a2.x, a2.y), pack2bf(a2.z, a2.w),
                   pack2bf(a3.x, a3.y), pack2bf(a3.z, a3.w));
  return __builtin_bit_cast(v16bf, r);
}

// ---------------- GEMM: C(MxN) = A(MxK) * W(NxK)^T ----------------
// Block = 128 threads = 4 waves; block tile 64x64, wave tile 16x64, K-step 32.
template <int A_IS_BF16, int C_IS_F32>
__global__ __launch_bounds__(128)
void wmma_gemm_nt(const void* __restrict__ Av, const float* __restrict__ W,
                  void* __restrict__ Cv, int M, int N, int K) {
  int lane = threadIdx.x & 31;
  int wave = threadIdx.x >> 5;
  int m0 = blockIdx.y * 64 + wave * 16;
  int n0 = blockIdx.x * 64;
  v8f acc[4] = {};
  for (int kk = 0; kk < K; kk += 32) {
    v16bf a;
    if (A_IS_BF16) a = frag_a_bf16((const uint16_t*)Av, K, m0, kk);
    else           a = frag_a_f32((const float*)Av, K, m0, kk);
    if (kk + 32 < K) {  // speculative prefetch of next weight tile
      const float* prow = W + (size_t)(n0 + (lane & 15)) * K + kk + 32;
      __builtin_prefetch(prow, 0, 1);
    }
#pragma unroll
    for (int t = 0; t < 4; t++) {
      v16bf b = frag_b_f32(W, K, n0 + t * 16, kk);
      acc[t] = wmma_bf16(a, b, acc[t]);
    }
  }
  // C layout: lane L, VGPR r -> (M = r + 8*(L/16), N = L%16)
#pragma unroll
  for (int t = 0; t < 4; t++) {
    int n = n0 + t * 16 + (lane & 15);
#pragma unroll
    for (int r = 0; r < 8; r++) {
      int m = m0 + r + ((lane >> 4) << 3);
      float v = acc[t][r];
      if (C_IS_F32) ((float*)Cv)[(size_t)m * N + n] = v;
      else          ((uint16_t*)Cv)[(size_t)m * N + n] = f2bf(v);
    }
  }
}

// ---------------- RoPE (in place, bf16 storage, fp32 math) ----------------
__global__ __launch_bounds__(256)
void rope_kernel(uint16_t* __restrict__ Q, uint16_t* __restrict__ Kb,
                 const int* __restrict__ pos_ids) {
  int idx = blockIdx.x * blockDim.x + threadIdx.x;  // S*H*64 pairs
  if (idx >= S_LEN * NHEADS * 64) return;
  int j = idx & 63;
  int h = (idx >> 6) & (NHEADS - 1);
  int p = idx >> 11;
  float pos = (float)pos_ids[p];
  float inv = __powf(10000.0f, -(float)j * (1.0f / 64.0f));
  float ang = pos * inv;
  float c = __cosf(ang), s = __sinf(ang);
  size_t base = (size_t)p * DMODEL + h * HDIM + j;
  float q0 = bf2f(Q[base]), q1 = bf2f(Q[base + 64]);
  Q[base]      = f2bf(q0 * c - q1 * s);
  Q[base + 64] = f2bf(q1 * c + q0 * s);
  float k0 = bf2f(Kb[base]), k1 = bf2f(Kb[base + 64]);
  Kb[base]      = f2bf(k0 * c - k1 * s);
  Kb[base + 64] = f2bf(k1 * c + k0 * s);
}

__global__ void zero_kernel(float* __restrict__ p, int n) {
  int i = blockIdx.x * blockDim.x + threadIdx.x;
  if (i < n) p[i] = 0.0f;
}

// ---------------- Attention (flash 2-pass + H2O score accumulation) -------
// Grid (S/64, H); 4 waves per block, each wave owns 16 q-rows independently.
__global__ __launch_bounds__(128)
void attn_kernel(const uint16_t* __restrict__ Q, const uint16_t* __restrict__ Kc,
                 const uint16_t* __restrict__ V, uint16_t* __restrict__ Obf,
                 float* __restrict__ scores) {
  __shared__ __align__(16) uint16_t Vs[4][32][HDIM];  // per-wave V chunk
  __shared__ __align__(16) uint16_t Ps[4][16][32];    // per-wave P tile
  int lane = threadIdx.x & 31;
  int wave = threadIdx.x >> 5;
  int h = blockIdx.y;
  int q0 = blockIdx.x * 64 + wave * 16;
  const float scale = 0.088388347648318447f;  // 1/sqrt(128)

  // Q fragments (persist across all chunks)
  v16bf qf[4];
#pragma unroll
  for (int d = 0; d < 4; d++)
    qf[d] = frag_a_bf16(Q + (size_t)h * HDIM, DMODEL, q0, d * 32);

  float mrow[8], lrow[8];
#pragma unroll
  for (int r = 0; r < 8; r++) { mrow[r] = -1e30f; lrow[r] = 0.0f; }

  int nch = (q0 + 16 + 31) >> 5;  // causal: only chunks with kc <= q0+15

  // ---- pass A: row max / sum-exp ----
  for (int c = 0; c < nch; c++) {
    int kc = c * 32;
    v8f lg[2] = {};
#pragma unroll
    for (int t = 0; t < 2; t++)
#pragma unroll
      for (int d = 0; d < 4; d++) {
        v16bf kf = frag_b_bf16(Kc + (size_t)h * HDIM, DMODEL, kc + t * 16, d * 32);
        lg[t] = wmma_bf16(qf[d], kf, lg[t]);
      }
#pragma unroll
    for (int r = 0; r < 8; r++) {
      int m = q0 + r + ((lane >> 4) << 3);
      int k0i = kc + (lane & 15);
      float x0 = (k0i <= m) ? lg[0][r] * scale : -1e30f;
      float x1 = (k0i + 16 <= m) ? lg[1][r] * scale : -1e30f;
      float cm = fmaxf(x0, x1);
#pragma unroll
      for (int dd = 1; dd < 16; dd <<= 1) cm = fmaxf(cm, __shfl_xor(cm, dd, 32));
      float nm = fmaxf(mrow[r], cm);
      float es = __expf(x0 - nm) + __expf(x1 - nm);
#pragma unroll
      for (int dd = 1; dd < 16; dd <<= 1) es += __shfl_xor(es, dd, 32);
      lrow[r] = lrow[r] * __expf(mrow[r] - nm) + es;
      mrow[r] = nm;
    }
  }
  float rl[8];
#pragma unroll
  for (int r = 0; r < 8; r++) rl[r] = 1.0f / lrow[r];

  // ---- pass B: normalized P, scores, P@V ----
  v8f oacc[8] = {};
  for (int c = 0; c < nch; c++) {
    int kc = c * 32;
    { // Async-stage V chunk (32 keys x 128 bf16 = 8 KB) directly into LDS via
      // the CDNA5 async path (no VGPR round trip; tracked by ASYNCcnt).
      // WAR guard: async LDS writes are unordered vs prior ds_loads.
      asm volatile("s_wait_dscnt 0x0" ::: "memory");
      uint32_t ldsoff = (uint32_t)(uintptr_t)&Vs[wave][lane][0];
      const uint16_t* vr = V + (size_t)(kc + lane) * DMODEL + h * HDIM;
#pragma unroll
      for (int u = 0; u < 16; u++)
        asm volatile("global_load_async_to_lds_b128 %0, %1, off offset:%2"
                     :: "v"(ldsoff), "v"(vr), "i"(u * 16) : "memory");
    }
    v8f lg[2] = {};
#pragma unroll
    for (int t = 0; t < 2; t++)
#pragma unroll
      for (int d = 0; d < 4; d++) {
        v16bf kf = frag_b_bf16(Kc + (size_t)h * HDIM, DMODEL, kc + t * 16, d * 32);
        lg[t] = wmma_bf16(qf[d], kf, lg[t]);
      }
#pragma unroll
    for (int t = 0; t < 2; t++) {
      int k = kc + t * 16 + (lane & 15);
      float colsum = 0.0f;
#pragma unroll
      for (int r = 0; r < 8; r++) {
        int m = q0 + r + ((lane >> 4) << 3);
        float p = 0.0f;
        if (k <= m) p = __expf(lg[t][r] * scale - mrow[r]) * rl[r];
        colsum += p;
        Ps[wave][r + ((lane >> 4) << 3)][t * 16 + (lane & 15)] = f2bf(p);
      }
      colsum += __shfl_xor(colsum, 16, 32);
      if (lane < 16) atomicAdd(&scores[(size_t)h * S_LEN + k], colsum);
    }
    // P fragment (A layout) from LDS
    v16bf pf;
    {
      int lo = (lane >> 4) << 4;
      Frag32B r;
      r.a = *(const uint4*)&Ps[wave][lane & 15][lo];
      r.b = *(const uint4*)&Ps[wave][lane & 15][lo + 8];
      pf = __builtin_bit_cast(v16bf, r);
    }
    // Wait for the async V stage before gathering B fragments from LDS.
    asm volatile("s_wait_asynccnt 0x0" ::: "memory");
#pragma unroll
    for (int vt = 0; vt < 8; vt++) {
      uint32_t w[8];
#pragma unroll
      for (int i = 0; i < 8; i++) {
        uint16_t lo16 = Vs[wave][((lane >> 4) << 4) + 2 * i][vt * 16 + (lane & 15)];
        uint16_t hi16 = Vs[wave][((lane >> 4) << 4) + 2 * i + 1][vt * 16 + (lane & 15)];
        w[i] = (uint32_t)lo16 | ((uint32_t)hi16 << 16);
      }
      Frag32B r;
      r.a = make_uint4(w[0], w[1], w[2], w[3]);
      r.b = make_uint4(w[4], w[5], w[6], w[7]);
      oacc[vt] = wmma_bf16(pf, __builtin_bit_cast(v16bf, r), oacc[vt]);
    }
  }
  // store O tile as bf16 into attn_out [s][4096]
#pragma unroll
  for (int vt = 0; vt < 8; vt++) {
    int n = h * HDIM + vt * 16 + (lane & 15);
#pragma unroll
    for (int r = 0; r < 8; r++) {
      int m = q0 + r + ((lane >> 4) << 3);
      Obf[(size_t)m * DMODEL + n] = f2bf(oacc[vt][r]);
    }
  }
}

// ---------------- H2O top-k mask + previous_scores ----------------
__global__ __launch_bounds__(256)
void topk_kernel(const float* __restrict__ scores, float* __restrict__ outv) {
  float* mask_out = outv + (size_t)S_LEN * DMODEL;          // [32][2049]
  float* prev_out = mask_out + (size_t)NHEADS * MASKW;      // [32][2048]
  int h = blockIdx.x;
  int tid = threadIdx.x;
  __shared__ float vals[SELW];
  __shared__ float rv[256];
  __shared__ int   ri[256];
  const float* sc = scores + (size_t)h * S_LEN;
  float* mrow = mask_out + (size_t)h * MASKW;
  for (int i = tid; i < SELW; i += 256) vals[i] = sc[i];
  for (int i = tid; i < MASKW; i += 256) mrow[i] = (i >= SELW) ? 1.0f : 0.0f;
  __syncthreads();
  for (int it = 0; it < HEAVY; it++) {
    float bv = -1e30f; int bi = 0;
    for (int i = tid; i < SELW; i += 256) {
      float v = vals[i];
      if (v > bv) { bv = v; bi = i; }
    }
    rv[tid] = bv; ri[tid] = bi;
    __syncthreads();
    for (int off = 128; off > 0; off >>= 1) {
      if (tid < off && rv[tid + off] > rv[tid]) { rv[tid] = rv[tid + off]; ri[tid] = ri[tid + off]; }
      __syncthreads();
    }
    if (tid == 0) { int b = ri[0]; mrow[b] = 1.0f; vals[b] = -1e30f; }
    __syncthreads();
  }
  for (int i = tid; i < S_LEN; i += 256) {
    float mv = (i >= SELW) ? 1.0f : mrow[i];
    prev_out[(size_t)h * S_LEN + i] = sc[i] * mv;
  }
}

// ---------------- host-side orchestration ----------------
extern "C" void kernel_launch(void* const* d_in, const int* in_sizes, int n_in,
                              void* d_out, int out_size, void* d_ws, size_t ws_size,
                              hipStream_t stream) {
  (void)in_sizes; (void)n_in; (void)out_size; (void)ws_size;
  const float* X   = (const float*)d_in[0];
  // d_in[1] = attention_mask (causal; applied analytically)
  const int*   pos = (const int*)d_in[2];
  const float* Wq  = (const float*)d_in[3];
  const float* Wk  = (const float*)d_in[4];
  const float* Wv  = (const float*)d_in[5];
  const float* Wo  = (const float*)d_in[6];
  float* out = (float*)d_out;

  const size_t SD = (size_t)S_LEN * DMODEL;
  uint16_t* Qp = (uint16_t*)d_ws;          // bf16 [s][d]
  uint16_t* Kp = Qp + SD;
  uint16_t* Vp = Kp + SD;
  uint16_t* Ob = Vp + SD;                  // attention output, bf16
  float* scores = (float*)(Ob + SD);       // [32][2048] fp32

  dim3 gg(DMODEL / 64, S_LEN / 64);
  wmma_gemm_nt<0, 0><<<gg, 128, 0, stream>>>(X, Wq, Qp, S_LEN, DMODEL, DMODEL);
  wmma_gemm_nt<0, 0><<<gg, 128, 0, stream>>>(X, Wk, Kp, S_LEN, DMODEL, DMODEL);
  wmma_gemm_nt<0, 0><<<gg, 128, 0, stream>>>(X, Wv, Vp, S_LEN, DMODEL, DMODEL);

  int npairs = S_LEN * NHEADS * 64;
  rope_kernel<<<npairs / 256, 256, 0, stream>>>(Qp, Kp, pos);

  zero_kernel<<<(NHEADS * S_LEN + 255) / 256, 256, 0, stream>>>(scores, NHEADS * S_LEN);

  attn_kernel<<<dim3(S_LEN / 64, NHEADS), 128, 0, stream>>>(Qp, Kp, Vp, Ob, scores);

  wmma_gemm_nt<1, 1><<<gg, 128, 0, stream>>>(Ob, Wo, out, S_LEN, DMODEL, DMODEL);

  topk_kernel<<<NHEADS, 256, 0, stream>>>(scores, out);
}
